// MultiHeadLocalAttention_7078106104051
// MI455X (gfx1250) — matrix operand order repaired and checked by reading, not compiled, not run on hardware
//
#include <hip/hip_runtime.h>
#include <hip/hip_bf16.h>

#define EMB   512
#define HEADS 8
#define DHEAD 64
#define NQ    8192
#define NNB   32

typedef __attribute__((ext_vector_type(16))) __bf16 v16bf;
typedef __attribute__((ext_vector_type(8)))  float  v8f;

union Frag16 {
    v16bf v;
    uint4 q[2];
    unsigned int u32[8];
    unsigned short u16[16];
};

__device__ __forceinline__ unsigned short f32_to_bf16(float f) {
    unsigned int x = __float_as_uint(f);
    unsigned int r = (x + 0x7FFFu + ((x >> 16) & 1u)) >> 16;   // RNE
    return (unsigned short)r;
}

__device__ __forceinline__ float bf16lo_to_f32(unsigned int u) {
    return __uint_as_float(u << 16);
}
__device__ __forceinline__ float bf16hi_to_f32(unsigned int u) {
    return __uint_as_float(u & 0xFFFF0000u);
}

// ---------------- x: fp32 -> bf16, 4 elements/thread ----------------
__global__ void cvt_x_kernel(const float4* __restrict__ x4,
                             unsigned long long* __restrict__ xb4, int ntot4) {
    int i = blockIdx.x * blockDim.x + threadIdx.x;
    if (i < ntot4) {
        float4 f = x4[i];
        unsigned long long r =
              (unsigned long long)f32_to_bf16(f.x)
            | ((unsigned long long)f32_to_bf16(f.y) << 16)
            | ((unsigned long long)f32_to_bf16(f.z) << 32)
            | ((unsigned long long)f32_to_bf16(f.w) << 48);
        xb4[i] = r;
    }
}

// ---------------- pack W (fp32) directly into WMMA B-fragment layout ----------------
// Fragment f = ((w*32 + ntile)*16 + kstep): 32 lanes x 32B, total 3*512 frags = 1.5MB.
// grid = (32 ntiles, 16 ksteps, 3), block = 32.
__global__ void __launch_bounds__(32)
pack_w_kernel(const float* __restrict__ Wq,
              const float* __restrict__ Wk,
              const float* __restrict__ Wv,
              uint4* __restrict__ wpack) {
    const int lane = threadIdx.x & 31;
    const int half = lane >> 4;
    const int l16  = lane & 15;
    const int tileN = blockIdx.x * 16;
    const int kk    = blockIdx.y * 32;
    const int w     = blockIdx.z;
    const float* W = (w == 0) ? Wq : (w == 1) ? Wk : Wv;

    Frag16 b;
    #pragma unroll
    for (int t = 0; t < 8; ++t) {
        int kbr = kk + 16 * half + 2 * t;
        const float* Wp = W + (size_t)kbr * EMB + tileN + l16;
        b.u16[2 * t]     = f32_to_bf16(Wp[0]);
        b.u16[2 * t + 1] = f32_to_bf16(Wp[EMB]);
    }
    size_t frag = (size_t)w * 512 + (size_t)blockIdx.x * 16 + blockIdx.y;
    uint4* dst = wpack + frag * 64 + (size_t)lane * 2;   // 64 uint4 per frag
    dst[0] = b.q[0];
    dst[1] = b.q[1];
}

// ---------------- QKV projection GEMM (bf16 WMMA, f32 accum, LDS-staged B) ----------------
__device__ __forceinline__ void load_A(const unsigned short* __restrict__ Arow0,
                                       const unsigned short* __restrict__ Arow1,
                                       int ks, Frag16& a0, Frag16& a1) {
    a0.q[0] = *(const uint4*)(Arow0 + ks * 32);
    a0.q[1] = *(const uint4*)(Arow0 + ks * 32 + 8);
    a1.q[0] = *(const uint4*)(Arow1 + ks * 32);
    a1.q[1] = *(const uint4*)(Arow1 + ks * 32 + 8);
}

__device__ __forceinline__ void load_B(const uint4* Blds, int ks,
                                       Frag16& b0, Frag16& b1,
                                       Frag16& b2, Frag16& b3) {
    const uint4* Bk = Blds + ks * 64;
    b0.q[0] = Bk[0 * 16 * 64]; b0.q[1] = Bk[0 * 16 * 64 + 1];
    b1.q[0] = Bk[1 * 16 * 64]; b1.q[1] = Bk[1 * 16 * 64 + 1];
    b2.q[0] = Bk[2 * 16 * 64]; b2.q[1] = Bk[2 * 16 * 64 + 1];
    b3.q[0] = Bk[3 * 16 * 64]; b3.q[1] = Bk[3 * 16 * 64 + 1];
}

#define WMMA8(A0, A1, B0, B1, B2, B3)                                                                          \
    c00 = __builtin_amdgcn_wmma_f32_16x16x32_bf16(false, (A0).v, false, (B0).v, (short)0, c00, false, false);  \
    c10 = __builtin_amdgcn_wmma_f32_16x16x32_bf16(false, (A1).v, false, (B0).v, (short)0, c10, false, false);  \
    c01 = __builtin_amdgcn_wmma_f32_16x16x32_bf16(false, (A0).v, false, (B1).v, (short)0, c01, false, false);  \
    c11 = __builtin_amdgcn_wmma_f32_16x16x32_bf16(false, (A1).v, false, (B1).v, (short)0, c11, false, false);  \
    c02 = __builtin_amdgcn_wmma_f32_16x16x32_bf16(false, (A0).v, false, (B2).v, (short)0, c02, false, false);  \
    c12 = __builtin_amdgcn_wmma_f32_16x16x32_bf16(false, (A1).v, false, (B2).v, (short)0, c12, false, false);  \
    c03 = __builtin_amdgcn_wmma_f32_16x16x32_bf16(false, (A0).v, false, (B3).v, (short)0, c03, false, false);  \
    c13 = __builtin_amdgcn_wmma_f32_16x16x32_bf16(false, (A1).v, false, (B3).v, (short)0, c13, false, false);

// Block = 8 waves (256 threads). Each wave computes a 32x64 output tile (2 M-tiles x
// 4 N-tiles), so each LDS B fragment feeds 2 WMMAs. Block covers 256 rows x 64 cols.
// B strip (64 contiguous fragments = 64KB) staged in LDS once. 2-stage pipelined K-loop.
// Output written as bf16 (halves downstream gather traffic).
// grid = (EMB/64, NQ/256, 3), block = 256.
__global__ void __launch_bounds__(256)
qkv_gemm_kernel(const unsigned short* __restrict__ xb,   // 8192x512 bf16 row-major
                const uint4* __restrict__ wpack,         // packed B fragments
                const float* __restrict__ bq,
                const float* __restrict__ bk,
                const float* __restrict__ bv,
                unsigned short* __restrict__ qkvb)       // 3 x (8192x512) bf16
{
    __shared__ uint4 lds_b[4096];                        // 64 frags x 64 uint4 = 64KB

    const int lane   = threadIdx.x & 31;
    const int wv     = threadIdx.x >> 5;                 // 0..7
    const int half   = lane >> 4;
    const int l16    = lane & 15;
    const int tileN  = blockIdx.x * 64;
    const int ntile0 = blockIdx.x * 4;
    const int tileM  = blockIdx.y * 256 + wv * 32;
    const int w      = blockIdx.z;

    const float*    bias = (w == 0) ? bq : (w == 1) ? bk : bv;
    unsigned short* out  = qkvb + (size_t)w * NQ * EMB;

    // ---- stage B strip into LDS (contiguous 64KB) ----
    const uint4* bsrc = wpack + ((size_t)w * 512 + (size_t)ntile0 * 16) * 64;
    #pragma unroll
    for (int i = 0; i < 16; ++i)
        lds_b[threadIdx.x + i * 256] = bsrc[threadIdx.x + i * 256];
    __syncthreads();

    const unsigned short* Arow0 = xb + (size_t)(tileM + l16) * EMB + 16 * half;
    const unsigned short* Arow1 = Arow0 + (size_t)16 * EMB;
    const uint4* Blds = lds_b + (size_t)lane * 2;

    v8f c00 = {}, c01 = {}, c02 = {}, c03 = {};
    v8f c10 = {}, c11 = {}, c12 = {}, c13 = {};

    Frag16 a0A, a1A, b0A, b1A, b2A, b3A;   // stage A (even ks)
    Frag16 a0B, a1B, b0B, b1B, b2B, b3B;   // stage B (odd ks)

    load_A(Arow0, Arow1, 0, a0A, a1A);  load_B(Blds, 0, b0A, b1A, b2A, b3A);
    load_A(Arow0, Arow1, 1, a0B, a1B);  load_B(Blds, 1, b0B, b1B, b2B, b3B);

    #pragma unroll 1
    for (int ks = 0; ks < 14; ks += 2) {
        __builtin_prefetch(Arow0 + (ks + 2) * 32, 0, 3);
        __builtin_prefetch(Arow1 + (ks + 2) * 32, 0, 3);
        WMMA8(a0A, a1A, b0A, b1A, b2A, b3A);                       // consume stage ks
        load_A(Arow0, Arow1, ks + 2, a0A, a1A);
        load_B(Blds, ks + 2, b0A, b1A, b2A, b3A);
        WMMA8(a0B, a1B, b0B, b1B, b2B, b3B);                       // consume stage ks+1
        load_A(Arow0, Arow1, ks + 3, a0B, a1B);
        load_B(Blds, ks + 3, b0B, b1B, b2B, b3B);
    }
    WMMA8(a0A, a1A, b0A, b1A, b2A, b3A);                           // ks = 14
    WMMA8(a0B, a1B, b0B, b1B, b2B, b3B);                           // ks = 15

    // C/D layout: VGPR r -> row (r + 8*half), col = l16 within each subtile
    #pragma unroll
    for (int s = 0; s < 4; ++s) {
        const v8f& ca = (s == 0) ? c00 : (s == 1) ? c01 : (s == 2) ? c02 : c03;
        const v8f& cb = (s == 0) ? c10 : (s == 1) ? c11 : (s == 2) ? c12 : c13;
        const int col = tileN + s * 16 + l16;
        const float bbias = bias[col];
        #pragma unroll
        for (int r = 0; r < 8; ++r) {
            out[(size_t)(tileM + 8 * half + r) * EMB + col]      = f32_to_bf16(ca[r] + bbias);
            out[(size_t)(tileM + 16 + 8 * half + r) * EMB + col] = f32_to_bf16(cb[r] + bbias);
        }
    }
}

// ---------------- gathered local attention (bf16 Q/K/V, f32 math) ----------------
// One wave32 per (query, head). Lane owns d-elements {2*lane, 2*lane+1}.
__global__ void __launch_bounds__(256)
attn_kernel(const unsigned short* __restrict__ Qb,
            const unsigned short* __restrict__ Kb,
            const unsigned short* __restrict__ Vb,
            const int*   __restrict__ A,
            float* __restrict__ out,        // (8192 x 512) f32
            float* __restrict__ attsum)     // (8192 x 8)
{
    const int lane = threadIdx.x & 31;
    const int wave = blockIdx.x * (blockDim.x >> 5) + (threadIdx.x >> 5);
    const int q    = wave >> 3;
    const int head = wave & (HEADS - 1);
    const int dofs = head * DHEAD + 2 * lane;

    const unsigned int uq = *(const unsigned int*)(Qb + (size_t)q * EMB + dofs);
    const float qx = bf16lo_to_f32(uq), qy = bf16hi_to_f32(uq);
    const int myIdx = A[q * NNB + lane];   // lane j owns neighbor j

    // Pass 1: energies e_j = q . k[idx_j]
    float myE = 0.0f;
    #pragma unroll 8
    for (int j = 0; j < NNB; ++j) {
        int kidx = __shfl(myIdx, j, 32);
        unsigned int uk = *(const unsigned int*)(Kb + (size_t)kidx * EMB + dofs);
        float p = qx * bf16lo_to_f32(uk) + qy * bf16hi_to_f32(uk);
        p += __shfl_xor(p, 1, 32);
        p += __shfl_xor(p, 2, 32);
        p += __shfl_xor(p, 4, 32);
        p += __shfl_xor(p, 8, 32);
        p += __shfl_xor(p, 16, 32);
        if (lane == j) myE = p;
    }

    // softmax over 32 neighbors (one value per lane)
    float m = myE;
    m = fmaxf(m, __shfl_xor(m, 1, 32));
    m = fmaxf(m, __shfl_xor(m, 2, 32));
    m = fmaxf(m, __shfl_xor(m, 4, 32));
    m = fmaxf(m, __shfl_xor(m, 8, 32));
    m = fmaxf(m, __shfl_xor(m, 16, 32));
    float wgt = __expf(myE - m);
    float s = wgt;
    s += __shfl_xor(s, 1, 32);
    s += __shfl_xor(s, 2, 32);
    s += __shfl_xor(s, 4, 32);
    s += __shfl_xor(s, 8, 32);
    s += __shfl_xor(s, 16, 32);
    const float scaling = 22.627416997969522f;   // sqrt(512)
    const float att = wgt / (s * scaling);

    // Pass 2: out = sum_j att_j * v[idx_j]
    float ox = 0.0f, oy = 0.0f;
    #pragma unroll 8
    for (int j = 0; j < NNB; ++j) {
        float aj  = __shfl(att, j, 32);
        int   vix = __shfl(myIdx, j, 32);
        unsigned int uv = *(const unsigned int*)(Vb + (size_t)vix * EMB + dofs);
        ox += aj * bf16lo_to_f32(uv);
        oy += aj * bf16hi_to_f32(uv);
    }
    float2 o; o.x = ox; o.y = oy;
    *(float2*)(out + (size_t)q * EMB + dofs) = o;

    float as = att;
    as += __shfl_xor(as, 1, 32);
    as += __shfl_xor(as, 2, 32);
    as += __shfl_xor(as, 4, 32);
    as += __shfl_xor(as, 8, 32);
    as += __shfl_xor(as, 16, 32);
    if (lane == 0) attsum[q * HEADS + head] = as;
}

__global__ void att_reduce_kernel(const float* __restrict__ attsum,
                                  float* __restrict__ Att) {
    int q = blockIdx.x * blockDim.x + threadIdx.x;
    if (q < NQ) {
        float s = 0.0f;
        #pragma unroll
        for (int h = 0; h < HEADS; ++h) s += attsum[q * HEADS + h];
        Att[q] = s;
    }
}

extern "C" void kernel_launch(void* const* d_in, const int* in_sizes, int n_in,
                              void* d_out, int out_size, void* d_ws, size_t ws_size,
                              hipStream_t stream) {
    const float* x   = (const float*)d_in[0];
    const int*   A   = (const int*)  d_in[1];
    const float* Wq  = (const float*)d_in[2];
    const float* bq  = (const float*)d_in[3];
    const float* Wk  = (const float*)d_in[4];
    const float* bk  = (const float*)d_in[5];
    const float* Wv  = (const float*)d_in[6];
    const float* bv  = (const float*)d_in[7];

    char* ws = (char*)d_ws;
    unsigned short* xb     = (unsigned short*)(ws);                            //  8.0 MB
    uint4*          wpack  = (uint4*)         (ws + (size_t)8  * 1024 * 1024); //  1.5 MB
    unsigned short* qkvb   = (unsigned short*)(ws + (size_t)10 * 1024 * 1024); // 24.0 MB
    float*          attsum = (float*)         (ws + (size_t)34 * 1024 * 1024); //  0.25 MB

    float* out = (float*)d_out;
    float* Att = out + (size_t)NQ * EMB;

    cvt_x_kernel<<<(NQ * EMB / 4 + 255) / 256, 256, 0, stream>>>(
        (const float4*)x, (unsigned long long*)xb, NQ * EMB / 4);
    pack_w_kernel<<<dim3(EMB / 16, EMB / 32, 3), 32, 0, stream>>>(Wq, Wk, Wv, wpack);

    qkv_gemm_kernel<<<dim3(EMB / 64, NQ / 256, 3), 256, 0, stream>>>(
        xb, wpack, bq, bk, bv, qkvb);

    attn_kernel<<<NQ * HEADS / 8, 256, 0, stream>>>(
        qkvb,                        // Q
        qkvb + (size_t)NQ * EMB,     // K
        qkvb + (size_t)2 * NQ * EMB, // V
        A, out, attsum);

    att_reduce_kernel<<<NQ / 256, 256, 0, stream>>>(attsum, Att);
}
